// Attention_38689065402477
// MI455X (gfx1250) — matrix operand order repaired and checked by reading, not compiled, not run on hardware
//
#include <hip/hip_runtime.h>
#include <hip/hip_bf16.h>
#include <math.h>

// Problem constants (from reference): B=4, Lq=Lk=1024, D=1024, H=16, Dh=64
#define BSZ   4
#define LSEQ  1024
#define DMOD  1024
#define NHEAD 16
#define DHEAD 64
#define MROWS (BSZ * LSEQ)   // 4096

typedef __attribute__((ext_vector_type(16))) __bf16 v16bf;
typedef __attribute__((ext_vector_type(8)))  __bf16 v8bf;
typedef __attribute__((ext_vector_type(8)))  float  v8f;
typedef __attribute__((ext_vector_type(4)))  float  v4f;

static __device__ __forceinline__ v8f wmma_bf16(v16bf a, v16bf b, v8f c) {
    // D = A(16x32 bf16) x B(32x16 bf16) + C(16x16 f32)
    return __builtin_amdgcn_wmma_f32_16x16x32_bf16(
        /*neg_a=*/false, a, /*neg_b=*/false, b,
        /*c_mod=*/(short)0, c, /*reuse_a=*/false, /*reuse_b=*/false);
}

// Max-reduce across a 16-lane DPP row (lanes 0-15 and 16-31 independently)
// using a rotate butterfly: after ror 1,2,4,8 every lane holds the row max.
static __device__ __forceinline__ float row_reduce_max16(float v) {
    int t;
    t = __builtin_amdgcn_update_dpp(0, __float_as_int(v), 0x121, 0xf, 0xf, true); // row_ror:1
    v = fmaxf(v, __int_as_float(t));
    t = __builtin_amdgcn_update_dpp(0, __float_as_int(v), 0x122, 0xf, 0xf, true); // row_ror:2
    v = fmaxf(v, __int_as_float(t));
    t = __builtin_amdgcn_update_dpp(0, __float_as_int(v), 0x124, 0xf, 0xf, true); // row_ror:4
    v = fmaxf(v, __int_as_float(t));
    t = __builtin_amdgcn_update_dpp(0, __float_as_int(v), 0x128, 0xf, 0xf, true); // row_ror:8
    v = fmaxf(v, __int_as_float(t));
    return v;
}

// Async copy of one 16B chunk from global to LDS (per-lane addresses).
static __device__ __forceinline__ void async_copy_b128(unsigned lds_off, const void* gaddr) {
    asm volatile("global_load_async_to_lds_b128 %0, %1, off"
                 :: "v"(lds_off), "v"(gaddr) : "memory");
}

// ---------------------------------------------------------------------------
// Kernel 1: transpose + fp32->bf16 convert a [K=1024][N=1024] weight into
// WT[n][k] so GEMM B-fragments are contiguous along k.
// ---------------------------------------------------------------------------
__global__ __launch_bounds__(256) void wT_kernel(const float* __restrict__ W,
                                                 __bf16* __restrict__ WT) {
    __shared__ float tile[32][33];
    const int tx = threadIdx.x & 31;   // 0..31
    const int ty = threadIdx.x >> 5;   // 0..7
    const int n0 = blockIdx.x * 32;
    const int k0 = blockIdx.y * 32;
#pragma unroll
    for (int i = 0; i < 4; ++i) {
        int k = ty + i * 8;
        tile[k][tx] = W[(size_t)(k0 + k) * DMOD + n0 + tx];
    }
    __syncthreads();
#pragma unroll
    for (int i = 0; i < 4; ++i) {
        int n = ty + i * 8;
        WT[(size_t)(n0 + n) * DMOD + k0 + tx] = (__bf16)tile[tx][n];
    }
}

// ---------------------------------------------------------------------------
// Kernel 2: projection GEMM  C[M=4096,N=1024] = relu(A_f32 @ W + bias)
// A read as fp32 and converted to bf16 in-register; W consumed via WT (bf16).
// 8 waves/block; wave tile = 32(M) x 64(N) = 8 WMMA C-tiles.
// TOUT=false: store bf16 row-major [M,D].  TOUT=true: store transposed [D,M]
// (used for V so attention's PV B-fragments are contiguous).
// ---------------------------------------------------------------------------
template <bool TOUT>
__global__ __launch_bounds__(256) void proj_kernel(const float* __restrict__ A,
                                                   const __bf16* __restrict__ WT,
                                                   const float* __restrict__ bias,
                                                   __bf16* __restrict__ Cout) {
    const int lane = threadIdx.x & 31;
    const int wave = threadIdx.x >> 5;   // 0..7
    const int wm = wave >> 2;            // 0..1
    const int wn = wave & 3;             // 0..3
    const int m_base = blockIdx.y * 64 + wm * 32;
    const int n_base = blockIdx.x * 256 + wn * 64;
    const int mrow = lane & 15;
    const int hi   = lane >> 4;          // 0 or 1

    v8f acc[2][4] = {};

    for (int k0 = 0; k0 < DMOD; k0 += 32) {
        // Prefetch next k-slice (global_prefetch_b8; speculative, dropped at end)
        __builtin_prefetch(A + (size_t)(m_base + mrow) * DMOD + k0 + 32, 0, 1);
        __builtin_prefetch(WT + (size_t)(n_base + mrow) * DMOD + k0 + 32, 0, 1);

        // A-fragments (16x32): elem i -> K = hi*16 + i
        v16bf af[2];
#pragma unroll
        for (int mt = 0; mt < 2; ++mt) {
            const float* pr = A + (size_t)(m_base + mt * 16 + mrow) * DMOD + k0 + hi * 16;
            v4f x0 = *(const v4f*)(pr);
            v4f x1 = *(const v4f*)(pr + 4);
            v4f x2 = *(const v4f*)(pr + 8);
            v4f x3 = *(const v4f*)(pr + 12);
            v16bf a;
#pragma unroll
            for (int e = 0; e < 4; ++e) {
                a[e]      = (__bf16)x0[e];
                a[4 + e]  = (__bf16)x1[e];
                a[8 + e]  = (__bf16)x2[e];
                a[12 + e] = (__bf16)x3[e];
            }
            af[mt] = a;
        }
        // B-fragments (32x16): elem i -> K = hi*16 + i, column = mrow
        v16bf bfr[4];
#pragma unroll
        for (int nt = 0; nt < 4; ++nt) {
            const __bf16* pb = WT + (size_t)(n_base + nt * 16 + mrow) * DMOD + k0 + hi * 16;
            bfr[nt] = *(const v16bf*)pb;   // 32B contiguous
        }
#pragma unroll
        for (int mt = 0; mt < 2; ++mt)
#pragma unroll
            for (int nt = 0; nt < 4; ++nt)
                acc[mt][nt] = wmma_bf16(af[mt], bfr[nt], acc[mt][nt]);
    }

    // Epilogue: bias + ReLU + bf16 store
#pragma unroll
    for (int mt = 0; mt < 2; ++mt) {
#pragma unroll
        for (int nt = 0; nt < 4; ++nt) {
            const int n = n_base + nt * 16 + mrow;   // this lane's column
            const float bn = bias[n];
            if (!TOUT) {
#pragma unroll
                for (int j = 0; j < 8; ++j) {        // row = j + 8*hi in tile
                    int r = m_base + mt * 16 + hi * 8 + j;
                    float v = acc[mt][nt][j] + bn;
                    v = v > 0.0f ? v : 0.0f;
                    Cout[(size_t)r * DMOD + n] = (__bf16)v;
                }
            } else {
                v8bf pk;
#pragma unroll
                for (int j = 0; j < 8; ++j) {
                    float v = acc[mt][nt][j] + bn;
                    v = v > 0.0f ? v : 0.0f;
                    pk[j] = (__bf16)v;
                }
                // Transposed: consecutive rows are contiguous -> 16B store
                *(v8bf*)(Cout + (size_t)n * MROWS + m_base + mt * 16 + hi * 8) = pk;
            }
        }
    }
}

// ---------------------------------------------------------------------------
// Kernel 3: flash attention per (batch, head). All 8 waves of a block share
// (b,h), so K/V 32-key tiles are staged ONCE per block into LDS with double-
// buffered async copies (global_load_async_to_lds_b128 / ASYNCcnt), then
// consumed as WMMA B-fragments from LDS. Softmax: DPP row-max butterfly +
// ones-column WMMA denominator. grid = (B*H, Lq/128), block = 256 (8 waves).
// ---------------------------------------------------------------------------
__global__ __launch_bounds__(256) void attn_kernel(const __bf16* __restrict__ Qb,
                                                   const __bf16* __restrict__ Kb,
                                                   const __bf16* __restrict__ Vt,
                                                   const float* __restrict__ Xq,
                                                   float* __restrict__ Out) {
    __shared__ __bf16 kTile[2][32][64];  // [buf][key][dh]   2 x 4 KB
    __shared__ __bf16 vTile[2][64][32];  // [buf][dh][key]   2 x 4 KB
    __shared__ __bf16 pLds[8][16][32];   // per-wave P slab  8 KB
    const int tid  = threadIdx.x;
    const int lane = tid & 31;
    const int wave = tid >> 5;
    const int mrow = lane & 15;
    const int hi   = lane >> 4;
    const int bh = blockIdx.x;           // 0..63
    const int b = bh >> 4;
    const int h = bh & 15;
    const int q0 = blockIdx.y * 128 + wave * 16;

    // Per-thread async-copy assignments (one 16B chunk of K, one of V)
    const int krow = tid >> 3, kch = tid & 7;           // K: 32 rows x 8 chunks
    const int vrow = tid >> 2, vch = tid & 3;           // V: 64 rows x 4 chunks
    const __bf16* gK = Kb + (size_t)(b * LSEQ + krow) * DMOD + h * DHEAD + kch * 8;
    const __bf16* gV = Vt + (size_t)(h * DHEAD + vrow) * MROWS + b * LSEQ + vch * 8;

    auto issue_tile = [&](int buf, int kb) {
        unsigned lk = (unsigned)(size_t)&kTile[buf][krow][kch * 8];
        unsigned lv = (unsigned)(size_t)&vTile[buf][vrow][vch * 8];
        async_copy_b128(lk, gK + (size_t)kb * DMOD);   // advance K by kb rows
        async_copy_b128(lv, gV + kb);                  // advance V by kb cols
    };

    // Q A-fragments for Dh chunks [0,32) and [32,64)
    v16bf aq[2];
#pragma unroll
    for (int t = 0; t < 2; ++t) {
        const __bf16* pq = Qb + (size_t)(b * LSEQ + q0 + mrow) * DMOD + h * DHEAD + t * 32 + hi * 16;
        v8bf lo = *(const v8bf*)pq;
        v8bf hh = *(const v8bf*)(pq + 8);
        v16bf a;
#pragma unroll
        for (int e = 0; e < 8; ++e) { a[e] = lo[e]; a[8 + e] = hh[e]; }
        aq[t] = a;
    }

    // All-ones B fragment: acc_l = P @ ones gives row-sums of P in C layout.
    v16bf ones;
#pragma unroll
    for (int e = 0; e < 16; ++e) ones[e] = (__bf16)1.0f;

    v8f acc[4] = {};
    v8f accl = {};                        // softmax denominator accumulator
    float mrun[8];
#pragma unroll
    for (int j = 0; j < 8; ++j) mrun[j] = -1e30f;

    issue_tile(0, 0);                     // prime the pipeline
    int cur = 0;

    for (int kb = 0; kb < LSEQ; kb += 32) {
        const bool more = (kb + 32) < LSEQ;
        if (more) issue_tile(cur ^ 1, kb + 32);       // DMA next tile
        // Async loads complete in order: waiting down to the 2 just-issued
        // guarantees the current buffer's 2 loads have landed.
        if (more) asm volatile("s_wait_asynccnt 0x2" ::: "memory");
        else      asm volatile("s_wait_asynccnt 0x0" ::: "memory");
        __syncthreads();                               // current tile visible to all waves

        // ---- S = Q @ K^T  (two 16x16 S tiles: keys kb..+15, kb+16..+31) ----
        v8f s[2] = {};
#pragma unroll
        for (int t2 = 0; t2 < 2; ++t2) {
#pragma unroll
            for (int t = 0; t < 2; ++t) {
                const __bf16* pk = &kTile[cur][t2 * 16 + mrow][t * 32 + hi * 16];
                v16bf bk = *(const v16bf*)pk;   // B[k,n] = K[key=n, dh=k]
                s[t2] = wmma_bf16(aq[t], bk, s[t2]);
            }
        }
        // ---- online softmax over the 32 new columns ----
#pragma unroll
        for (int j = 0; j < 8; ++j) {
            float v0 = s[0][j], v1 = s[1][j];
            float mx = row_reduce_max16(fmaxf(v0, v1));      // DPP butterfly
            float mnew  = fmaxf(mrun[j], mx);
            float scale = __expf(mrun[j] - mnew);
            float p0 = __expf(v0 - mnew);
            float p1 = __expf(v1 - mnew);
            mrun[j] = mnew;
#pragma unroll
            for (int t = 0; t < 4; ++t) acc[t][j] *= scale;
            accl[j] *= scale;
            // C layout: this lane holds row (j + 8*hi), cols mrow / 16+mrow
            pLds[wave][hi * 8 + j][mrow]      = (__bf16)p0;
            pLds[wave][hi * 8 + j][16 + mrow] = (__bf16)p1;
        }
        asm volatile("s_wait_dscnt 0x0" ::: "memory");   // intra-wave cross-lane P exchange

        // ---- re-load P in A-fragment layout ----
        const __bf16* pl = &pLds[wave][mrow][hi * 16];
        v8bf plo = *(const v8bf*)pl;
        v8bf phi = *(const v8bf*)(pl + 8);
        v16bf pa;
#pragma unroll
        for (int e = 0; e < 8; ++e) { pa[e] = plo[e]; pa[8 + e] = phi[e]; }

        // ---- denominator: accl += P @ ones (row sums, no cross-lane ops) ----
        accl = wmma_bf16(pa, ones, accl);

        // ---- ctx += P(16x32) @ V(32x64) ----
#pragma unroll
        for (int t = 0; t < 4; ++t) {
            const __bf16* pv = &vTile[cur][t * 16 + mrow][hi * 16];
            v16bf bv = *(const v16bf*)pv;   // B[k,n] = V[key=k, dh=n] via V^T
            acc[t] = wmma_bf16(pa, bv, acc[t]);
        }

        __syncthreads();   // all waves done with 'cur' before it is re-filled
        cur ^= 1;
    }

    // ---- epilogue: normalize, fuse residual 0.5*q + 0.5*ctx ----
#pragma unroll
    for (int j = 0; j < 8; ++j) {
        const float inv = 1.0f / accl[j];
        const int r = q0 + hi * 8 + j;
#pragma unroll
        for (int t = 0; t < 4; ++t) {
            int c = h * DHEAD + t * 16 + mrow;
            size_t idx = (size_t)(b * LSEQ + r) * DMOD + c;
            Out[idx] = 0.5f * Xq[idx] + 0.5f * (acc[t][j] * inv);
        }
    }
}

// ---------------------------------------------------------------------------
extern "C" void kernel_launch(void* const* d_in, const int* in_sizes, int n_in,
                              void* d_out, int out_size, void* d_ws, size_t ws_size,
                              hipStream_t stream) {
    (void)in_sizes; (void)n_in; (void)out_size; (void)ws_size;
    const float* queries = (const float*)d_in[0];
    const float* keys    = (const float*)d_in[1];
    const float* Wq = (const float*)d_in[2];
    const float* bq = (const float*)d_in[3];
    const float* Wk = (const float*)d_in[4];
    const float* bk = (const float*)d_in[5];
    const float* Wv = (const float*)d_in[6];
    const float* bv = (const float*)d_in[7];
    float* out = (float*)d_out;

    char* ws = (char*)d_ws;
    __bf16* WqT = (__bf16*)(ws + ((size_t)0  << 20));   // 2 MB
    __bf16* WkT = (__bf16*)(ws + ((size_t)2  << 20));   // 2 MB
    __bf16* WvT = (__bf16*)(ws + ((size_t)4  << 20));   // 2 MB
    __bf16* Qb  = (__bf16*)(ws + ((size_t)6  << 20));   // 8 MB  [4096,1024]
    __bf16* Kb  = (__bf16*)(ws + ((size_t)14 << 20));   // 8 MB  [4096,1024]
    __bf16* Vt  = (__bf16*)(ws + ((size_t)22 << 20));   // 8 MB  [1024,4096]

    dim3 tb(256);
    dim3 gT(32, 32);
    wT_kernel<<<gT, tb, 0, stream>>>(Wq, WqT);
    wT_kernel<<<gT, tb, 0, stream>>>(Wk, WkT);
    wT_kernel<<<gT, tb, 0, stream>>>(Wv, WvT);

    dim3 gG(DMOD / 256, MROWS / 64);                    // (4, 64)
    proj_kernel<false><<<gG, tb, 0, stream>>>(queries, WqT, bq, Qb);
    proj_kernel<false><<<gG, tb, 0, stream>>>(keys,    WkT, bk, Kb);
    proj_kernel<true ><<<gG, tb, 0, stream>>>(keys,    WvT, bv, Vt);

    dim3 gA(BSZ * NHEAD, LSEQ / 128);                   // (64, 8)
    attn_kernel<<<gA, tb, 0, stream>>>(Qb, Kb, Vt, queries, out);
}
